// MultiHeadAttention_26182120637059
// MI455X (gfx1250) — matrix operand order repaired and checked by reading, not compiled, not run on hardware
//
#include <hip/hip_runtime.h>

// ---------------------------------------------------------------------------
// MHA forward for MI455X (gfx1250, wave32, WMMA).
// bf16 inputs / fp32 accumulate on v_wmma_f32_16x16x32_bf16.
// Async global->LDS staging (ASYNCcnt) with double-buffered K/V tiles.
// ---------------------------------------------------------------------------

typedef __attribute__((ext_vector_type(16))) __bf16 v16bf;
typedef __attribute__((ext_vector_type(8)))  __bf16 v8bf;
typedef __attribute__((ext_vector_type(8)))  float  v8f;
typedef __attribute__((ext_vector_type(4)))  float  f32x4;
typedef __attribute__((ext_vector_type(4)))  unsigned int u32x4;
typedef __attribute__((ext_vector_type(2)))  unsigned int u32x2;
typedef __attribute__((ext_vector_type(4)))  int    i32x4;

#define NUM_HEADS 16
#define HEAD_DIM  64
#define SEQ       2048
#define DIM       1024
#define INNER     1024
#define SCALE_F   0.125f   // 64^-0.5

// ---- CDNA5 async global->LDS copy (ASYNCcnt), with portable fallback -------
#if defined(__has_builtin)
#if __has_builtin(__builtin_amdgcn_global_load_async_to_lds_b128)
#define HAVE_ASYNC_COPY 1
#endif
#endif
#ifndef HAVE_ASYNC_COPY
#define HAVE_ASYNC_COPY 0
#endif

__device__ __forceinline__ void async_copy16(const void* gsrc, void* ldst) {
#if HAVE_ASYNC_COPY
  typedef __attribute__((address_space(1))) i32x4 gvec_t;   // global v4i
  typedef __attribute__((address_space(3))) i32x4 lvec_t;   // LDS v4i
  gvec_t* g = (gvec_t*)(unsigned long long)gsrc;
  lvec_t* l = (lvec_t*)(unsigned int)(unsigned long long)ldst;
  __builtin_amdgcn_global_load_async_to_lds_b128(g, l, 0, 0);
#else
  *(u32x4*)ldst = *(const u32x4*)gsrc;
#endif
}

// wait until all async copies from this wave have landed in LDS
__device__ __forceinline__ void async_join0() {
#if HAVE_ASYNC_COPY
#if __has_builtin(__builtin_amdgcn_s_wait_asynccnt)
  __builtin_amdgcn_s_wait_asynccnt(0);
#else
  asm volatile("s_wait_asynccnt 0x0" ::: "memory");
#endif
#endif
}

// wait until <=8 outstanding: with 8 copies per stage and in-order completion,
// this guarantees the previous stage's tile is resident while the newest
// stage's 8 copies remain in flight (latency hiding behind WMMA work).
__device__ __forceinline__ void async_join8() {
#if HAVE_ASYNC_COPY
#if __has_builtin(__builtin_amdgcn_s_wait_asynccnt)
  __builtin_amdgcn_s_wait_asynccnt(8);
#else
  asm volatile("s_wait_asynccnt 0x8" ::: "memory");
#endif
#endif
}

__device__ __forceinline__ v8f vzero8() {
  v8f v;
#pragma unroll
  for (int i = 0; i < 8; ++i) v[i] = 0.0f;
  return v;
}

union BfPack16 { u32x4 u; v8bf h; };   // 16 bytes = 8 bf16
union BfPack8  { u32x2 u; __bf16 h[4]; };

// A fragment, 16x32 bf16 (MxK). Lane l: row M=l&15; K = {0..7,16..23} (+8 for
// lanes 16-31). Source must be row-major [M][K-contig], 16B aligned.
__device__ __forceinline__ v16bf frag_a_ld(const __bf16* base, int stride, int lane) {
  const int m  = lane & 15;
  const int kb = (lane >> 4) << 3;
  const __bf16* p = base + m * stride + kb;
  BfPack16 lo, hi;
  lo.u = *(const u32x4*)(p);
  hi.u = *(const u32x4*)(p + 16);
  v16bf f;
#pragma unroll
  for (int i = 0; i < 8; ++i) { f[i] = lo.h[i]; f[i + 8] = hi.h[i]; }
  return f;
}

// B fragment, 32x16 bf16 (KxN). Lane l: col N=l&15; K = {0..7,16..23} (+8 for
// lanes 16-31). Source must be [N][K-contig], 16B aligned.
__device__ __forceinline__ v16bf frag_b_ld(const __bf16* base, int stride, int lane) {
  const int n  = lane & 15;
  const int kb = (lane >> 4) << 3;
  const __bf16* p = base + n * stride + kb;
  BfPack16 lo, hi;
  lo.u = *(const u32x4*)(p);
  hi.u = *(const u32x4*)(p + 16);
  v16bf f;
#pragma unroll
  for (int i = 0; i < 8; ++i) { f[i] = lo.h[i]; f[i + 8] = hi.h[i]; }
  return f;
}

__device__ __forceinline__ v8f wmma_bf16(v16bf a, v16bf b, v8f c) {
  return __builtin_amdgcn_wmma_f32_16x16x32_bf16(
      false, a, false, b, (short)0, c, false, false);
}

// ---------------------------------------------------------------------------
// Kernel 1: QKV projection.  C[4096,3072] = x[4096,1024] @ [Wq;Wkv]^T
// Scatters: Q,K -> [B,H,S,64] bf16 ; V -> [B,H,64,S] bf16 (transposed).
// ---------------------------------------------------------------------------
__global__ __launch_bounds__(256) void qkv_proj_kernel(
    const float* __restrict__ x, const float* __restrict__ Wq,
    const float* __restrict__ Wkv,
    unsigned short* __restrict__ Qb_, unsigned short* __restrict__ Kb_,
    unsigned short* __restrict__ Vt_) {
  __bf16* Qb = (__bf16*)Qb_;
  __bf16* Kb = (__bf16*)Kb_;
  __bf16* Vt = (__bf16*)Vt_;
  __shared__ __align__(16) __bf16 As[128][32];
  __shared__ __align__(16) __bf16 Bs[128][32];
  const int tid  = threadIdx.x;
  const int lane = tid & 31;
  const int wave = tid >> 5;
  const int wm = wave >> 2;          // 0..1 -> 64 rows
  const int wn = wave & 3;           // 0..3 -> 32 cols
  const int bm = blockIdx.x * 128;
  const int bn = blockIdx.y * 128;   // 0..23 tiles over N=3072
  const int which = bn >> 10;        // 0=q 1=k 2=v (block-uniform)
  const float* Wbase = (which == 0) ? (Wq + (size_t)bn * DIM)
                                    : (Wkv + (size_t)(bn - 1024) * DIM);
  v8f acc[4][2];
#pragma unroll
  for (int mi = 0; mi < 4; ++mi)
#pragma unroll
    for (int ni = 0; ni < 2; ++ni) acc[mi][ni] = vzero8();

  for (int k0 = 0; k0 < DIM; k0 += 32) {
    if (k0 + 32 < DIM) {   // hint next tile into cache (global_prefetch_b8)
      const int row0 = tid >> 3, col0 = (tid & 7) << 2;
      __builtin_prefetch(x + (size_t)(bm + row0) * DIM + k0 + 32 + col0, 0, 3);
      __builtin_prefetch(Wbase + (size_t)row0 * DIM + k0 + 32 + col0, 0, 3);
    }
#pragma unroll
    for (int i = 0; i < 4; ++i) {          // 128x32 f32 -> bf16 stage, both tiles
      const int c   = tid + i * 256;       // 0..1023
      const int row = c >> 3;
      const int col = (c & 7) << 2;
      f32x4 fa = *(const f32x4*)(x + (size_t)(bm + row) * DIM + k0 + col);
      f32x4 fb = *(const f32x4*)(Wbase + (size_t)row * DIM + k0 + col);
      BfPack8 ca, cb;
      ca.h[0] = (__bf16)fa.x; ca.h[1] = (__bf16)fa.y;
      ca.h[2] = (__bf16)fa.z; ca.h[3] = (__bf16)fa.w;
      cb.h[0] = (__bf16)fb.x; cb.h[1] = (__bf16)fb.y;
      cb.h[2] = (__bf16)fb.z; cb.h[3] = (__bf16)fb.w;
      *(u32x2*)&As[row][col] = ca.u;
      *(u32x2*)&Bs[row][col] = cb.u;
    }
    __syncthreads();
    v16bf a[4], b[2];
#pragma unroll
    for (int mi = 0; mi < 4; ++mi)
      a[mi] = frag_a_ld(&As[wm * 64 + mi * 16][0], 32, lane);
#pragma unroll
    for (int ni = 0; ni < 2; ++ni)
      b[ni] = frag_b_ld(&Bs[wn * 32 + ni * 16][0], 32, lane);
#pragma unroll
    for (int mi = 0; mi < 4; ++mi)
#pragma unroll
      for (int ni = 0; ni < 2; ++ni)
        acc[mi][ni] = wmma_bf16(a[mi], b[ni], acc[mi][ni]);
    __syncthreads();
  }

  // epilogue: block-uniform q/k/v selection, packed stores for the V path
  const int group = lane >> 4;
  const int ln    = lane & 15;
  __bf16* dstQK = (which == 1) ? Kb : Qb;
#pragma unroll
  for (int mi = 0; mi < 4; ++mi) {
    const int gm0 = bm + wm * 64 + mi * 16 + (group << 3);
    const int bb  = gm0 >> 11;
    const int s0  = gm0 & 2047;
#pragma unroll
    for (int ni = 0; ni < 2; ++ni) {
      const int gn = bn + wn * 32 + ni * 16 + ln;
      const int h  = (gn & 1023) >> 6;
      const int d  = gn & 63;
      if (which < 2) {
        __bf16* p = dstQK + (((size_t)bb * NUM_HEADS + h) * SEQ + s0) * HEAD_DIM + d;
#pragma unroll
        for (int r = 0; r < 8; ++r)
          p[(size_t)r * HEAD_DIM] = (__bf16)acc[mi][ni][r];
      } else {
        // V^T: 8 consecutive s per lane -> one 16B store
        __bf16* p = Vt + (((size_t)bb * NUM_HEADS + h) * HEAD_DIM + d) * SEQ + s0;
        BfPack16 pk;
#pragma unroll
        for (int r = 0; r < 8; ++r) pk.h[r] = (__bf16)acc[mi][ni][r];
        *(u32x4*)p = pk.u;
      }
    }
  }
}

// ---------------------------------------------------------------------------
// Kernel 2: flash attention.  grid = (S/128, B*H); 8 waves, each wave owns
// 16 query rows x full 128-key width (row stats reduce inside 16-lane group).
// K/V tiles double-buffered via GLOBAL_LOAD_ASYNC_TO_LDS_B128:
// next tile's copies fly behind the current tile's WMMA + softmax work.
// ---------------------------------------------------------------------------
__global__ __launch_bounds__(256) void attn_kernel(
    const unsigned short* __restrict__ Qb_, const unsigned short* __restrict__ Kb_,
    const unsigned short* __restrict__ Vt_, unsigned short* __restrict__ AO_) {
  const __bf16* Qb = (const __bf16*)Qb_;
  const __bf16* Kb = (const __bf16*)Kb_;
  const __bf16* Vt = (const __bf16*)Vt_;
  __bf16* AO = (__bf16*)AO_;

  __shared__ __align__(16) __bf16 Ks[2][128][64];    // [buf][key][d]
  __shared__ __align__(16) __bf16 Vs[2][64][128];    // [buf][d][key]  (V^T)
  __shared__ __align__(16) __bf16 Ps[8][16][128];    // per-wave P tile

  const int tid   = threadIdx.x;
  const int lane  = tid & 31;
  const int wave  = tid >> 5;
  const int group = lane >> 4;
  const int ln    = lane & 15;
  const int qt    = blockIdx.x;                    // 0..15
  const int bh    = blockIdx.y;                    // 0..31
  const size_t headOff = (size_t)bh * SEQ * HEAD_DIM;
  const size_t vBase   = (size_t)bh * HEAD_DIM * SEQ;
  const int q0 = qt * 128 + wave * 16;

  // stage one K/V tile (8 async b128 copies per thread)
  auto stage = [&](int buf, int kt) {
    const __bf16* ksrc = Kb + headOff + (size_t)(kt + (tid >> 1)) * HEAD_DIM
                         + (tid & 1) * 32;
    __bf16* kdst = &Ks[buf][tid >> 1][(tid & 1) * 32];
#pragma unroll
    for (int i = 0; i < 4; ++i) async_copy16(ksrc + i * 8, kdst + i * 8);
    const __bf16* vsrc = Vt + vBase + (size_t)(tid >> 2) * SEQ + kt
                         + (tid & 3) * 32;
    __bf16* vdst = &Vs[buf][tid >> 2][(tid & 3) * 32];
#pragma unroll
    for (int i = 0; i < 4; ++i) async_copy16(vsrc + i * 8, vdst + i * 8);
  };

  // Q A-fragments held in registers for the whole kernel (K-dim = 64 -> 2 frags)
  v16bf qa[2];
  {
    const __bf16* qrow = Qb + headOff + (size_t)(q0 + ln) * HEAD_DIM;
    const int kb = group << 3;
#pragma unroll
    for (int c = 0; c < 2; ++c) {
      const __bf16* p = qrow + c * 32 + kb;
      BfPack16 lo, hi;
      lo.u = *(const u32x4*)(p);
      hi.u = *(const u32x4*)(p + 16);
#pragma unroll
      for (int i = 0; i < 8; ++i) { qa[c][i] = lo.h[i]; qa[c][i + 8] = hi.h[i]; }
    }
  }

  float mrow[8], lrow[8];
  v8f acc[4];
#pragma unroll
  for (int r = 0; r < 8; ++r) { mrow[r] = -1e30f; lrow[r] = 0.0f; }
#pragma unroll
  for (int di = 0; di < 4; ++di) acc[di] = vzero8();

  stage(0, 0);                       // prologue: first tile in flight

  const int NT = SEQ / 128;          // 16 tiles
  for (int it = 0; it < NT; ++it) {
    const int cur = it & 1;
    if (it + 1 < NT) {
      stage(cur ^ 1, (it + 1) * 128);  // issue next tile before waiting
      async_join8();                   // oldest 8 (== current tile) resident
    } else {
      async_join0();
    }
    __syncthreads();

    // S = (Q K^T) * scale    (16 q-rows x 128 keys per wave)
    v8f sf[8];
#pragma unroll
    for (int ni = 0; ni < 8; ++ni) {
      v8f s = vzero8();
      s = wmma_bf16(qa[0], frag_b_ld(&Ks[cur][ni * 16][0], 64, lane), s);
      s = wmma_bf16(qa[1], frag_b_ld(&Ks[cur][ni * 16][32], 64, lane), s);
#pragma unroll
      for (int r = 0; r < 8; ++r) s[r] *= SCALE_F;
      sf[ni] = s;
    }

    // online softmax per row (row lives in one 16-lane group)
#pragma unroll
    for (int r = 0; r < 8; ++r) {
      float mx = mrow[r];
#pragma unroll
      for (int ni = 0; ni < 8; ++ni) mx = fmaxf(mx, sf[ni][r]);
      for (int off = 1; off < 16; off <<= 1)
        mx = fmaxf(mx, __shfl_xor(mx, off, 32));
      const float corr = __expf(mrow[r] - mx);
      mrow[r] = mx;
      float sum = 0.0f;
#pragma unroll
      for (int ni = 0; ni < 8; ++ni) {
        const float p = __expf(sf[ni][r] - mx);
        sf[ni][r] = p;
        sum += p;
      }
      for (int off = 1; off < 16; off <<= 1) sum += __shfl_xor(sum, off, 32);
      lrow[r] = lrow[r] * corr + sum;
#pragma unroll
      for (int di = 0; di < 4; ++di) acc[di][r] *= corr;
    }

    // P -> per-wave LDS tile (C-layout -> A-layout transpose through LDS)
#pragma unroll
    for (int ni = 0; ni < 8; ++ni)
#pragma unroll
      for (int r = 0; r < 8; ++r)
        Ps[wave][r + (group << 3)][ni * 16 + ln] = (__bf16)sf[ni][r];

    // acc += P @ V
#pragma unroll
    for (int kc = 0; kc < 4; ++kc) {
      v16bf pa = frag_a_ld(&Ps[wave][0][kc * 32], 128, lane);
#pragma unroll
      for (int di = 0; di < 4; ++di)
        acc[di] = wmma_bf16(pa,
                            frag_b_ld(&Vs[cur][di * 16][kc * 32], 128, lane),
                            acc[di]);
    }
    __syncthreads();   // protects next iteration's async writes into buf cur
  }

  // normalize + store attn output as [B,S,H*64] bf16
  const int bb = bh >> 4, h = bh & 15;
#pragma unroll
  for (int r = 0; r < 8; ++r) {
    const float inv = 1.0f / lrow[r];
    const int q = q0 + r + (group << 3);
#pragma unroll
    for (int di = 0; di < 4; ++di) {
      const int d = di * 16 + ln;
      AO[((size_t)(bb * SEQ + q)) * INNER + h * HEAD_DIM + d] =
          (__bf16)(acc[di][r] * inv);
    }
  }
}

// ---------------------------------------------------------------------------
// Kernel 3: output projection + bias.  out[4096,1024] = AO @ Wout^T + bout
// A tile (already bf16) staged via async LDS copy; W converted in VGPRs.
// ---------------------------------------------------------------------------
__global__ __launch_bounds__(256) void out_proj_kernel(
    const unsigned short* __restrict__ AO_, const float* __restrict__ Wout,
    const float* __restrict__ bout, float* __restrict__ out) {
  const __bf16* AO = (const __bf16*)AO_;
  __shared__ __align__(16) __bf16 As[128][32];
  __shared__ __align__(16) __bf16 Bs[128][32];
  const int tid  = threadIdx.x;
  const int lane = tid & 31;
  const int wave = tid >> 5;
  const int wm = wave >> 2, wn = wave & 3;
  const int bm = blockIdx.x * 128;
  const int bn = blockIdx.y * 128;
  v8f acc[4][2];
#pragma unroll
  for (int mi = 0; mi < 4; ++mi)
#pragma unroll
    for (int ni = 0; ni < 2; ++ni) acc[mi][ni] = vzero8();

  for (int k0 = 0; k0 < INNER; k0 += 32) {
    if (k0 + 32 < INNER) {
      const int row0 = tid >> 3, col0 = (tid & 7) << 2;
      __builtin_prefetch(Wout + (size_t)(bn + row0) * INNER + k0 + 32 + col0, 0, 3);
    }
    {   // A already bf16: async copy straight into LDS
      const int row = tid >> 1, col = (tid & 1) * 16;
      const __bf16* src = AO + (size_t)(bm + row) * INNER + k0 + col;
      __bf16* dst = &As[row][col];
      async_copy16(src, dst);
      async_copy16(src + 8, dst + 8);
    }
#pragma unroll
    for (int i = 0; i < 4; ++i) {   // B: f32 -> bf16
      const int c   = tid + i * 256;
      const int row = c >> 3;
      const int col = (c & 7) << 2;
      f32x4 fb = *(const f32x4*)(Wout + (size_t)(bn + row) * INNER + k0 + col);
      BfPack8 cb;
      cb.h[0] = (__bf16)fb.x; cb.h[1] = (__bf16)fb.y;
      cb.h[2] = (__bf16)fb.z; cb.h[3] = (__bf16)fb.w;
      *(u32x2*)&Bs[row][col] = cb.u;
    }
    async_join0();
    __syncthreads();
    v16bf a[4], b[2];
#pragma unroll
    for (int mi = 0; mi < 4; ++mi)
      a[mi] = frag_a_ld(&As[wm * 64 + mi * 16][0], 32, lane);
#pragma unroll
    for (int ni = 0; ni < 2; ++ni)
      b[ni] = frag_b_ld(&Bs[wn * 32 + ni * 16][0], 32, lane);
#pragma unroll
    for (int mi = 0; mi < 4; ++mi)
#pragma unroll
      for (int ni = 0; ni < 2; ++ni)
        acc[mi][ni] = wmma_bf16(a[mi], b[ni], acc[mi][ni]);
    __syncthreads();
  }

  const int group = lane >> 4;
  const int ln    = lane & 15;
#pragma unroll
  for (int ni = 0; ni < 2; ++ni) {
    const int gn = bn + wn * 32 + ni * 16 + ln;
    const float bv = bout[gn];
#pragma unroll
    for (int mi = 0; mi < 4; ++mi) {
#pragma unroll
      for (int r = 0; r < 8; ++r) {
        const int gm = bm + wm * 64 + mi * 16 + r + (group << 3);
        out[(size_t)gm * DIM + gn] = acc[mi][ni][r] + bv;
      }
    }
  }
}

// ---------------------------------------------------------------------------
extern "C" void kernel_launch(void* const* d_in, const int* in_sizes, int n_in,
                              void* d_out, int out_size, void* d_ws, size_t ws_size,
                              hipStream_t stream) {
  (void)in_sizes; (void)n_in; (void)out_size; (void)ws_size;
  const float* x    = (const float*)d_in[0];
  const float* Wq   = (const float*)d_in[1];
  const float* Wkv  = (const float*)d_in[2];
  const float* Wout = (const float*)d_in[3];
  const float* bout = (const float*)d_in[4];
  float* out = (float*)d_out;

  // workspace layout (bf16): Q | K | V^T | attn_out  = 4 x 8MB = 32MB
  char* ws = (char*)d_ws;
  unsigned short* Qb = (unsigned short*)(ws);
  unsigned short* Kb = (unsigned short*)(ws + (size_t)8 * 1024 * 1024);
  unsigned short* Vt = (unsigned short*)(ws + (size_t)16 * 1024 * 1024);
  unsigned short* AO = (unsigned short*)(ws + (size_t)24 * 1024 * 1024);

  dim3 blk(256);
  qkv_proj_kernel<<<dim3(32, 24), blk, 0, stream>>>(x, Wq, Wkv, Qb, Kb, Vt);
  attn_kernel<<<dim3(16, 32), blk, 0, stream>>>(Qb, Kb, Vt, AO);
  out_proj_kernel<<<dim3(32, 8), blk, 0, stream>>>(AO, Wout, bout, out);
}
